// GCN_1116691497401
// MI455X (gfx1250) — compile-verified, matrix operand order checked by reading, not run on hardware
//
#include <hip/hip_runtime.h>
#include <hip/hip_bf16.h>
#include <stdint.h>

typedef __attribute__((ext_vector_type(16))) _Float16 v16h;
typedef __attribute__((ext_vector_type(8)))  _Float16 v8h;
typedef __attribute__((ext_vector_type(8)))  float    v8f;
typedef __attribute__((ext_vector_type(4)))  unsigned int v4u;
typedef __attribute__((ext_vector_type(8)))  int      v8i;
typedef __attribute__((ext_vector_type(4)))  int      v4i;

#define N_NODES 100000
#define N_EDGES 3200000
#define D_IN    512
#define D_HID   64
#define BN_EPS  1e-5f

// ---------------------------------------------------------------------------
// Elementwise / graph-prep kernels
// ---------------------------------------------------------------------------
__global__ void k_init(float* __restrict__ deg, float* __restrict__ agg1,
                       float* __restrict__ agg2) {
  long long t = blockIdx.x * 256LL + threadIdx.x;
  if (t < (long long)N_NODES * D_HID) {
    agg1[t] = 0.0f;
    agg2[t] = 0.0f;
    if (t < N_NODES) deg[t] = 1.0f;  // self-loop
  }
}

__global__ void k_degree(const int* __restrict__ dst, float* __restrict__ deg) {
  int e = blockIdx.x * 256 + threadIdx.x;
  if (e < N_EDGES) atomicAdd(&deg[dst[e]], 1.0f);
}

__global__ void k_rsqrt(float* __restrict__ deg) {
  int i = blockIdx.x * 256 + threadIdx.x;
  if (i < N_NODES) deg[i] = rsqrtf(deg[i]);  // deg buffer becomes dinv
}

__global__ void k_norm(const int* __restrict__ src, const int* __restrict__ dst,
                       const float* __restrict__ dinv, float* __restrict__ ne) {
  int e = blockIdx.x * 256 + threadIdx.x;
  if (e < N_EDGES) ne[e] = dinv[src[e]] * dinv[dst[e]];
}

// Pack a [K x 64] f32 weight matrix into WMMA B-fragment order (f16):
// out[((ksg*4 + nt)*32 + lane)*16 + i] = B[k][n] with
// k = ksg*32 + (i>>3)*16 + (lane>>4)*8 + (i&7), n = nt*16 + (lane&15)
__global__ void k_pack(const float* __restrict__ W, _Float16* __restrict__ out,
                       int ksteps) {
  int t = blockIdx.x * 256 + threadIdx.x;
  if (t >= ksteps * 128) return;
  int lane = t & 31;
  int nt = (t >> 5) & 3;
  int ksg = t >> 7;
  int hk = (lane >> 4) * 8;
  int n = nt * 16 + (lane & 15);
  _Float16* o = out + (size_t)t * 16;
#pragma unroll
  for (int i = 0; i < 16; ++i) {
    int k = ksg * 32 + (i >> 3) * 16 + hk + (i & 7);
    o[i] = (_Float16)W[(size_t)k * D_HID + n];
  }
}

// ---------------------------------------------------------------------------
// Tensor Data Mover: stage a [128 x 64] f32 tile (row stride D_IN) into LDS.
// tensor_dim1 = rows_rem => hardware zero-fills the M tail.
// ---------------------------------------------------------------------------
__device__ __forceinline__ void tdm_load_tile(const float* gsrc,
                                              unsigned lds_off, int rows_rem) {
  unsigned long long ga = (unsigned long long)(uintptr_t)gsrc;
  unsigned rr = rows_rem > 0 ? (unsigned)rows_rem : 0u;
  v4u g0;
  g0.x = 1u;                                            // count=1 valid descriptor
  g0.y = lds_off;                                       // lds_addr (bytes)
  g0.z = (unsigned)ga;                                  // global_addr[31:0]
  g0.w = ((unsigned)(ga >> 32) & 0x1FFFFFFu) | (2u << 30);  // addr[56:32] | type=2
  v8i g1;
  g1[0] = 0x00020000;                                   // data_size = 4B (code 2)
  g1[1] = (int)((unsigned)D_IN << 16);                  // tensor_dim0 = 512
  g1[2] = (int)((rr & 0xFFFFu) << 16);                  // tensor_dim1 lo16
  g1[3] = (int)(((rr >> 16) & 0xFFFFu) | (64u << 16));  // tensor_dim1 hi | tile_dim0=64
  g1[4] = 128;                                          // tile_dim1 = 128 rows
  g1[5] = D_IN;                                         // tensor_dim0_stride = 512
  g1[6] = 0;
  g1[7] = 0;
  v4i gz = {0, 0, 0, 0};
#if __clang_major__ >= 23
  v8i gz8 = {0, 0, 0, 0, 0, 0, 0, 0};
  __builtin_amdgcn_tensor_load_to_lds(g0, g1, gz, gz, gz8, 0);
#else
  __builtin_amdgcn_tensor_load_to_lds(g0, g1, gz, gz, 0);
#endif
}

// ---------------------------------------------------------------------------
// GEMM1: Y[N,64] = X[N,512](f32->f16) @ Bp(packed W1 f16), WMMA f32 accum.
// 256 threads = 8 waves; each wave owns a 16-row M-tile, full N=64.
// ---------------------------------------------------------------------------
__global__ __launch_bounds__(256) void k_gemm1(const float* __restrict__ X,
                                               const _Float16* __restrict__ Bp,
                                               float* __restrict__ Y) {
  __shared__ float tile[128 * 64];  // 32 KB
  const int tid = threadIdx.x;
  const int wave = tid >> 5;
  const int lane = tid & 31;
  const int lm = lane & 15;
  const int hk = (lane >> 4) * 8;
  const int mbase = blockIdx.x * 128;
  v8f acc0 = {}, acc1 = {}, acc2 = {}, acc3 = {};

  for (int kc = 0; kc < D_IN; kc += 64) {
    __syncthreads();  // previous tile fully consumed
    if (wave == 0) {
      tdm_load_tile(X + (size_t)mbase * D_IN + kc,
                    (unsigned)(uintptr_t)(void*)tile, N_NODES - mbase);
      __builtin_amdgcn_s_wait_tensorcnt(0);
    }
    __syncthreads();  // tile visible to all waves

    const float* arow = &tile[(wave * 16 + lm) * 64];
#pragma unroll
    for (int ks = 0; ks < 2; ++ks) {
      v16h a;
      const int kb = ks * 32 + hk;
#pragma unroll
      for (int j = 0; j < 8; ++j) {
        a[j]     = (_Float16)arow[kb + j];
        a[8 + j] = (_Float16)arow[kb + 16 + j];
      }
      const int ksg = (kc >> 5) + ks;
      const _Float16* bb = Bp + ((size_t)ksg * 4) * 32 * 16 + (size_t)lane * 16;
      v16h b0 = *(const v16h*)(bb);
      v16h b1 = *(const v16h*)(bb + 1 * 32 * 16);
      v16h b2 = *(const v16h*)(bb + 2 * 32 * 16);
      v16h b3 = *(const v16h*)(bb + 3 * 32 * 16);
      acc0 = __builtin_amdgcn_wmma_f32_16x16x32_f16(false, a, false, b0, (short)0, acc0, false, false);
      acc1 = __builtin_amdgcn_wmma_f32_16x16x32_f16(false, a, false, b1, (short)0, acc1, false, false);
      acc2 = __builtin_amdgcn_wmma_f32_16x16x32_f16(false, a, false, b2, (short)0, acc2, false, false);
      acc3 = __builtin_amdgcn_wmma_f32_16x16x32_f16(false, a, false, b3, (short)0, acc3, false, false);
    }
  }

  const int mrow0 = mbase + wave * 16 + (lane >> 4) * 8;
#pragma unroll
  for (int r = 0; r < 8; ++r) {
    const int m = mrow0 + r;
    if (m < N_NODES) {
      float* yr = Y + (size_t)m * D_HID + lm;
      yr[0]  = acc0[r];
      yr[16] = acc1[r];
      yr[32] = acc2[r];
      yr[48] = acc3[r];
    }
  }
}

// ---------------------------------------------------------------------------
// GEMM2: Y[N,64] = H[N,64](f16) @ Bp(packed W2 f16)
// ---------------------------------------------------------------------------
__global__ __launch_bounds__(256) void k_gemm2(const _Float16* __restrict__ H,
                                               const _Float16* __restrict__ Bp,
                                               float* __restrict__ Y) {
  const int tid = threadIdx.x;
  const int wave = tid >> 5;
  const int lane = tid & 31;
  const int lm = lane & 15;
  const int hk = (lane >> 4) * 8;
  const int mbase = blockIdx.x * 128;
  const int m = mbase + wave * 16 + lm;
  v8f acc0 = {}, acc1 = {}, acc2 = {}, acc3 = {};

#pragma unroll
  for (int ks = 0; ks < 2; ++ks) {
    v16h a;
    if (m < N_NODES) {
      const _Float16* hr = H + (size_t)m * D_HID + ks * 32 + hk;
      v8h lo = *(const v8h*)hr;
      v8h hi = *(const v8h*)(hr + 16);
#pragma unroll
      for (int j = 0; j < 8; ++j) { a[j] = lo[j]; a[8 + j] = hi[j]; }
    } else {
#pragma unroll
      for (int j = 0; j < 16; ++j) a[j] = (_Float16)0.0f;
    }
    const _Float16* bb = Bp + ((size_t)ks * 4) * 32 * 16 + (size_t)lane * 16;
    v16h b0 = *(const v16h*)(bb);
    v16h b1 = *(const v16h*)(bb + 1 * 32 * 16);
    v16h b2 = *(const v16h*)(bb + 2 * 32 * 16);
    v16h b3 = *(const v16h*)(bb + 3 * 32 * 16);
    acc0 = __builtin_amdgcn_wmma_f32_16x16x32_f16(false, a, false, b0, (short)0, acc0, false, false);
    acc1 = __builtin_amdgcn_wmma_f32_16x16x32_f16(false, a, false, b1, (short)0, acc1, false, false);
    acc2 = __builtin_amdgcn_wmma_f32_16x16x32_f16(false, a, false, b2, (short)0, acc2, false, false);
    acc3 = __builtin_amdgcn_wmma_f32_16x16x32_f16(false, a, false, b3, (short)0, acc3, false, false);
  }

  const int mrow0 = mbase + wave * 16 + (lane >> 4) * 8;
#pragma unroll
  for (int r = 0; r < 8; ++r) {
    const int mm = mrow0 + r;
    if (mm < N_NODES) {
      float* yr = Y + (size_t)mm * D_HID + lm;
      yr[0]  = acc0[r];
      yr[16] = acc1[r];
      yr[32] = acc2[r];
      yr[48] = acc3[r];
    }
  }
}

// ---------------------------------------------------------------------------
// Edge aggregation: agg[dst] += xw[src] * norm_e   (4 threads per edge)
// xw/agg are 25.6 MB -> L2-resident atomics.
// ---------------------------------------------------------------------------
__global__ void k_agg(const float* __restrict__ xw, const int* __restrict__ src,
                      const int* __restrict__ dst, const float* __restrict__ ne,
                      float* __restrict__ agg) {
  long long t = blockIdx.x * 256LL + threadIdx.x;
  if (t >= (long long)N_EDGES * 4) return;
  int e = (int)(t >> 2);
  int f = (int)(t & 3) * 16;
  float w = ne[e];
  const float* xs = xw + (size_t)src[e] * D_HID + f;
  float* ad = agg + (size_t)dst[e] * D_HID + f;
#pragma unroll
  for (int c = 0; c < 4; ++c) {
    float4 a = ((const float4*)xs)[c];
    atomicAdd(ad + 4 * c + 0, a.x * w);
    atomicAdd(ad + 4 * c + 1, a.y * w);
    atomicAdd(ad + 4 * c + 2, a.z * w);
    atomicAdd(ad + 4 * c + 3, a.w * w);
  }
}

// self-term + bias + BN(eval) + ReLU, output f16 for the next WMMA GEMM
__global__ void k_bnrelu(const float* __restrict__ xw, const float* __restrict__ agg,
                         const float* __restrict__ dinv, const float* __restrict__ b1,
                         const float* __restrict__ gamma, const float* __restrict__ beta,
                         const float* __restrict__ mean, const float* __restrict__ var,
                         _Float16* __restrict__ h16) {
  long long t = blockIdx.x * 256LL + threadIdx.x;
  if (t >= (long long)N_NODES * D_HID) return;
  int i = (int)(t >> 6);
  int f = (int)(t & 63);
  float di = dinv[i];
  float v = agg[t] + xw[t] * di * di + b1[f];
  v = gamma[f] * (v - mean[f]) * rsqrtf(var[f] + BN_EPS) + beta[f];
  h16[t] = (_Float16)fmaxf(v, 0.0f);
}

// self-term + bias (layer 2), f32 out
__global__ void k_self(const float* __restrict__ xw, const float* __restrict__ agg,
                       const float* __restrict__ dinv, const float* __restrict__ b2,
                       float* __restrict__ z) {
  long long t = blockIdx.x * 256LL + threadIdx.x;
  if (t >= (long long)N_NODES * D_HID) return;
  int i = (int)(t >> 6);
  int f = (int)(t & 63);
  float di = dinv[i];
  z[t] = agg[t] + xw[t] * di * di + b2[f];
}

// decode: out[e] = dot(z[src], z[dst]); 4 lanes/edge, shfl_xor quad reduce
__global__ void k_decode(const float* __restrict__ z, const int* __restrict__ src,
                         const int* __restrict__ dst, float* __restrict__ out) {
  long long t = blockIdx.x * 256LL + threadIdx.x;
  if (t >= (long long)N_EDGES * 4) return;
  int e = (int)(t >> 2);
  int q = (int)(t & 3);
  const float4* zs = (const float4*)(z + (size_t)src[e] * D_HID + q * 16);
  const float4* zd = (const float4*)(z + (size_t)dst[e] * D_HID + q * 16);
  float s = 0.0f;
#pragma unroll
  for (int c = 0; c < 4; ++c) {
    float4 a = zs[c], b = zd[c];
    s += a.x * b.x + a.y * b.y + a.z * b.z + a.w * b.w;
  }
  s += __shfl_xor(s, 1);
  s += __shfl_xor(s, 2);
  if (q == 0) out[e] = s;
}

// ---------------------------------------------------------------------------
extern "C" void kernel_launch(void* const* d_in, const int* in_sizes, int n_in,
                              void* d_out, int out_size, void* d_ws, size_t ws_size,
                              hipStream_t stream) {
  (void)in_sizes; (void)n_in; (void)out_size; (void)ws_size;
  const float* x     = (const float*)d_in[0];
  const int*   ei    = (const int*)d_in[1];
  const float* W1    = (const float*)d_in[2];
  const float* b1    = (const float*)d_in[3];
  const float* gamma = (const float*)d_in[4];
  const float* beta  = (const float*)d_in[5];
  const float* rmean = (const float*)d_in[6];
  const float* rvar  = (const float*)d_in[7];
  const float* W2    = (const float*)d_in[8];
  const float* b2    = (const float*)d_in[9];
  const int* src = ei;
  const int* dst = ei + N_EDGES;
  float* out = (float*)d_out;

  // workspace carve (256B aligned)
  char* w = (char*)d_ws;
  auto carve = [&](size_t bytes) {
    void* p = (void*)w;
    w += (bytes + 255) & ~(size_t)255;
    return p;
  };
  float*    deg  = (float*)carve((size_t)N_NODES * 4);            // -> dinv
  float*    ne   = (float*)carve((size_t)N_EDGES * 4);
  _Float16* w1p  = (_Float16*)carve((size_t)16 * 128 * 16 * 2);   // packed W1
  _Float16* w2p  = (_Float16*)carve((size_t)2 * 128 * 16 * 2);    // packed W2
  float*    xw   = (float*)carve((size_t)N_NODES * D_HID * 4);    // xw1 then xw2
  float*    aggA = (float*)carve((size_t)N_NODES * D_HID * 4);    // agg1 then z
  float*    aggB = (float*)carve((size_t)N_NODES * D_HID * 4);    // agg2
  _Float16* h16  = (_Float16*)carve((size_t)N_NODES * D_HID * 2);

  const int B = 256;
  const long long N64 = (long long)N_NODES * D_HID;
  const int gN64 = (int)((N64 + B - 1) / B);
  const int gE   = (N_EDGES + B - 1) / B;
  const int gE4  = (int)(((long long)N_EDGES * 4 + B - 1) / B);
  const int gN   = (N_NODES + B - 1) / B;
  const int gM   = (N_NODES + 127) / 128;

  k_init<<<gN64, B, 0, stream>>>(deg, aggA, aggB);
  k_degree<<<gE, B, 0, stream>>>(dst, deg);
  k_rsqrt<<<gN, B, 0, stream>>>(deg);
  k_norm<<<gE, B, 0, stream>>>(src, dst, deg, ne);
  k_pack<<<(16 * 128 + B - 1) / B, B, 0, stream>>>(W1, w1p, 16);
  k_pack<<<1, B, 0, stream>>>(W2, w2p, 2);

  k_gemm1<<<gM, B, 0, stream>>>(x, w1p, xw);
  k_agg<<<gE4, B, 0, stream>>>(xw, src, dst, ne, aggA);
  k_bnrelu<<<gN64, B, 0, stream>>>(xw, aggA, deg, b1, gamma, beta, rmean, rvar, h16);

  k_gemm2<<<gM, B, 0, stream>>>(h16, w2p, xw);
  k_agg<<<gE4, B, 0, stream>>>(xw, src, dst, ne, aggB);
  k_self<<<gN64, B, 0, stream>>>(xw, aggB, deg, b2, aggA);  // aggA reused as z

  k_decode<<<gE4, B, 0, stream>>>(aggA, src, dst, out);
}
